// TreeMLPEncoder_37838661878368
// MI455X (gfx1250) — compile-verified, matrix operand order checked by reading, not run on hardware
//
#include <hip/hip_runtime.h>
#include <hip/hip_bf16.h>

// ---------- types for WMMA / TDM ----------
typedef __attribute__((ext_vector_type(16))) __bf16        v16bf;
typedef __attribute__((ext_vector_type(8)))  float         v8f;
typedef __attribute__((ext_vector_type(4)))  unsigned int  u32x4;
typedef __attribute__((ext_vector_type(8)))  unsigned int  u32x8;

union Frag {
    v16bf v;
    u32x4 u[2];
};

__device__ __forceinline__ unsigned short f2bf(float x) {
    unsigned int u = __builtin_bit_cast(unsigned int, x);
    unsigned int r = (u + 0x7FFFu + ((u >> 16) & 1u)) >> 16;
    return (unsigned short)r;
}

__device__ __forceinline__ float gelu_exact(float x) {
    return 0.5f * x * (1.0f + erff(x * 0.70710678118654752f));
}

// TDM pads 4 dwords (8 bf16) after every 256 dwords (512 bf16) streamed.
// Map linear element offset -> padded LDS element offset.
__device__ __forceinline__ int lds_pad(int s) { return s + ((s >> 9) << 3); }

// ---------- Tensor Data Mover: 2-D bf16 tile (tile_d0 x tile_d1) -> LDS ----------
// D# per CDNA5 ISA. Padding: interval code 7 (256 dwords),
// amount code 3 (4 dwords). Issued once per wave; track with TENSORcnt.
__device__ __forceinline__ void tdm_load_2d(unsigned lds_off, const void* gptr,
                                            unsigned tile_d0, unsigned tile_d1,
                                            unsigned tensor_d0, unsigned tensor_d1,
                                            unsigned stride0_elems) {
    unsigned long long ga = (unsigned long long)gptr;
    unsigned galo = __builtin_amdgcn_readfirstlane((unsigned)ga);
    unsigned gahi = __builtin_amdgcn_readfirstlane((unsigned)(ga >> 32));

    u32x4 g0;
    g0[0] = 1u;                                   // count=1, no gather
    g0[1] = lds_off;                              // lds_addr (bytes)
    g0[2] = galo;                                 // global_addr[31:0]
    g0[3] = (gahi & 0x1FFFFFFu) | (2u << 30);     // global_addr[56:32] | type=2

    u32x8 g1;
    g1[0] = (1u << 16)        // data_size = 2 bytes
          | (1u << 20)        // pad_enable
          | (7u << 22)        // pad_interval: 256 dwords
          | (3u << 25);       // pad_amount:   4 dwords
    g1[1] = (tensor_d0 & 0xFFFFu) << 16;                       // tensor_dim0 lo16
    g1[2] = (tensor_d0 >> 16) | ((tensor_d1 & 0xFFFFu) << 16); // dim0 hi / dim1 lo
    g1[3] = (tensor_d1 >> 16) | (tile_d0 << 16);               // dim1 hi / tile_dim0
    g1[4] = tile_d1;                                           // tile_dim1 (tile_dim2=0)
    g1[5] = stride0_elems;                                     // tensor_dim0_stride lo32
    g1[6] = 0u;
    g1[7] = 0u;

    asm volatile("tensor_load_to_lds %0, %1" :: "s"(g0), "s"(g1) : "memory");
}

// ---------- E[n,d] = comp_table[component_ids[n], d] ----------
__global__ void init_embeds_kernel(float* __restrict__ E,
                                   const float* __restrict__ comp_table,
                                   const int* __restrict__ comp_ids) {
    int n = blockIdx.x;
    int d = threadIdx.x;
    int cid = comp_ids[n];
    E[(size_t)n * 256 + d] = comp_table[(size_t)cid * 256 + d];
}

// ---------- Wt[n,k] = bf16(W[k,n]);  W is [K,N] fp32 ----------
__global__ void cvt_transpose_kernel(const float* __restrict__ W,
                                     unsigned short* __restrict__ Wt,
                                     int K, int N) {
    int i = blockIdx.x * blockDim.x + threadIdx.x;
    if (i >= K * N) return;
    int n = i / K;
    int k = i - n * K;
    Wt[i] = f2bf(W[(size_t)k * N + n]);
}

// ---------- gather level inputs ----------
// X[row, 0:256]=op  [256:512]=le  [512:768]=re  [768:1024]=te (0 if no third)
__global__ void gather_kernel(const float* __restrict__ E,
                              const float* __restrict__ op_table,
                              const int* __restrict__ op_ids,
                              const int* __restrict__ lc,
                              const int* __restrict__ rc,
                              const int* __restrict__ tc,
                              const int* __restrict__ idx,
                              unsigned short* __restrict__ X,
                              float* __restrict__ Rb,
                              float* __restrict__ Rt,
                              int* __restrict__ H3) {
    int row = blockIdx.x;
    int d   = threadIdx.x;
    int node = idx[row];
    int l  = lc[node];
    int r  = rc[node];
    int t3 = tc[node];
    bool h = (t3 >= 0);
    int ts = h ? t3 : 0;

    float opv = op_table[(size_t)op_ids[node] * 256 + d];
    float le  = E[(size_t)l * 256 + d];
    float re  = E[(size_t)r * 256 + d];
    float te  = E[(size_t)ts * 256 + d] * (h ? 1.0f : 0.0f);

    size_t xb = (size_t)row * 1024;
    X[xb + d]        = f2bf(opv);
    X[xb + 256 + d]  = f2bf(le);
    X[xb + 512 + d]  = f2bf(re);
    X[xb + 768 + d]  = f2bf(te);

    Rb[(size_t)row * 256 + d] = le + re;
    Rt[(size_t)row * 256 + d] = (le + re + te) * (1.0f / 3.0f);
    if (d == 0) H3[row] = h ? 1 : 0;
}

// ---------- GEMM1 + bias + exact GELU: A[L,512] = gelu(X[:, :Kin] @ W1 + b1) ----------
// block = 256 threads (8 waves); block owns 32 rows (TDM-staged in LDS);
// wave w owns col tiles 4w..4w+3 for BOTH 16-row halves (B frags reused x2).
__global__ void __launch_bounds__(256)
mlp1_kernel(const unsigned short* __restrict__ X, int ldx, int Kin,
            const unsigned short* __restrict__ Wt,   // [512, Kin] bf16
            const float* __restrict__ bias,          // [512]
            unsigned short* __restrict__ A) {        // [L, 512] bf16
    __shared__ unsigned short sA[33280];             // 32*1024 + pads

    int lane = threadIdx.x & 31;
    int wave = threadIdx.x >> 5;
    int row0 = blockIdx.x * 32;
    int n0   = wave * 64;
    int m    = lane & 15;
    int ksel = (lane < 16) ? 0 : 8;

    if (wave == 0) {
        tdm_load_2d((unsigned)(size_t)sA,
                    X + (size_t)row0 * ldx,
                    (unsigned)Kin, 32u,
                    (unsigned)ldx, 32u,
                    (unsigned)ldx);
        __builtin_amdgcn_s_wait_tensorcnt(0);
    }
    __syncthreads();

    v8f acc[2][4] = {};

    for (int kb = 0; kb < Kin; kb += 32) {
        Frag a0, a1;
        int s0 = m * Kin + kb + ksel;
        int s1 = (m + 16) * Kin + kb + ksel;
        a0.u[0] = *(const u32x4*)(sA + lds_pad(s0));
        a0.u[1] = *(const u32x4*)(sA + lds_pad(s0 + 16));
        a1.u[0] = *(const u32x4*)(sA + lds_pad(s1));
        a1.u[1] = *(const u32x4*)(sA + lds_pad(s1 + 16));
#pragma unroll
        for (int t = 0; t < 4; ++t) {
            const unsigned short* bp = Wt + (size_t)(n0 + t * 16 + m) * Kin + kb + ksel;
            Frag b;
            b.u[0] = *(const u32x4*)(bp);
            b.u[1] = *(const u32x4*)(bp + 16);
            acc[0][t] = __builtin_amdgcn_wmma_f32_16x16x32_bf16(
                false, a0.v, false, b.v, (short)0, acc[0][t], false, false);
            acc[1][t] = __builtin_amdgcn_wmma_f32_16x16x32_bf16(
                false, a1.v, false, b.v, (short)0, acc[1][t], false, false);
        }
    }

    int rofs = (lane < 16) ? 0 : 8;
#pragma unroll
    for (int rh = 0; rh < 2; ++rh) {
#pragma unroll
        for (int t = 0; t < 4; ++t) {
#pragma unroll
            for (int i = 0; i < 8; ++i) {
                int rr = row0 + rh * 16 + i + rofs;
                int cc = n0 + t * 16 + m;
                float v = acc[rh][t][i] + bias[cc];
                A[(size_t)rr * 512 + cc] = f2bf(gelu_exact(v));
            }
        }
    }
}

// ---------- GEMM2 (both paths) + bias + residual + select + LayerNorm + scatter ----------
// block = 256 threads, one block per 16-row tile; Ab/At tiles TDM-staged in LDS;
// wave w owns col tiles 2w, 2w+1 of D=256.
__global__ void __launch_bounds__(256)
mlp2_kernel(const unsigned short* __restrict__ Ab,   // [L,512] bf16
            const unsigned short* __restrict__ At,   // [L,512] bf16
            const unsigned short* __restrict__ W2bT, // [256,512] bf16
            const unsigned short* __restrict__ W2tT, // [256,512] bf16
            const float* __restrict__ b2b,
            const float* __restrict__ b2t,
            const float* __restrict__ Rb,
            const float* __restrict__ Rt,
            const int* __restrict__ H3,
            const float* __restrict__ gamma,
            const float* __restrict__ beta,
            const int* __restrict__ idx,
            float* __restrict__ E) {
    __shared__ unsigned short sAb[8320];   // 16*512 + pads
    __shared__ unsigned short sAt[8320];
    __shared__ float sbuf[16][256];

    int lane = threadIdx.x & 31;
    int wave = threadIdx.x >> 5;
    int rb   = blockIdx.x * 16;
    int n0   = wave * 32;
    int m    = lane & 15;
    int ksel = (lane < 16) ? 0 : 8;

    if (wave == 0) {
        tdm_load_2d((unsigned)(size_t)sAb, Ab + (size_t)rb * 512,
                    512u, 16u, 512u, 16u, 512u);
        tdm_load_2d((unsigned)(size_t)sAt, At + (size_t)rb * 512,
                    512u, 16u, 512u, 16u, 512u);
        __builtin_amdgcn_s_wait_tensorcnt(0);
    }
    __syncthreads();

    v8f accb[2] = {};
    v8f acct[2] = {};

    for (int kb = 0; kb < 512; kb += 32) {
        Frag ab, at;
        int s = m * 512 + kb + ksel;
        ab.u[0] = *(const u32x4*)(sAb + lds_pad(s));
        ab.u[1] = *(const u32x4*)(sAb + lds_pad(s + 16));
        at.u[0] = *(const u32x4*)(sAt + lds_pad(s));
        at.u[1] = *(const u32x4*)(sAt + lds_pad(s + 16));
#pragma unroll
        for (int t = 0; t < 2; ++t) {
            const unsigned short* wb = W2bT + (size_t)(n0 + t * 16 + m) * 512 + kb + ksel;
            const unsigned short* wt = W2tT + (size_t)(n0 + t * 16 + m) * 512 + kb + ksel;
            Frag fb, ft;
            fb.u[0] = *(const u32x4*)(wb);
            fb.u[1] = *(const u32x4*)(wb + 16);
            ft.u[0] = *(const u32x4*)(wt);
            ft.u[1] = *(const u32x4*)(wt + 16);
            accb[t] = __builtin_amdgcn_wmma_f32_16x16x32_bf16(
                false, ab.v, false, fb.v, (short)0, accb[t], false, false);
            acct[t] = __builtin_amdgcn_wmma_f32_16x16x32_bf16(
                false, at.v, false, ft.v, (short)0, acct[t], false, false);
        }
    }

    int rofs = (lane < 16) ? 0 : 8;
#pragma unroll
    for (int t = 0; t < 2; ++t) {
#pragma unroll
        for (int i = 0; i < 8; ++i) {
            int rr   = i + rofs;          // local row 0..15
            int grow = rb + rr;
            int cc   = n0 + t * 16 + m;
            bool h = (H3[grow] != 0);
            float v;
            if (h) v = acct[t][i] + b2t[cc] + Rt[(size_t)grow * 256 + cc];
            else   v = accb[t][i] + b2b[cc] + Rb[(size_t)grow * 256 + cc];
            sbuf[rr][cc] = v;
        }
    }
    __syncthreads();

    // LayerNorm: wave w handles local rows 2w, 2w+1 (32 lanes x 8 cols each)
#pragma unroll
    for (int rsel = 0; rsel < 2; ++rsel) {
        int rr = wave * 2 + rsel;
        float s = 0.0f, s2 = 0.0f;
#pragma unroll
        for (int j = 0; j < 8; ++j) {
            float v = sbuf[rr][lane + 32 * j];
            s  += v;
            s2 += v * v;
        }
#pragma unroll
        for (int off = 16; off > 0; off >>= 1) {
            s  += __shfl_xor(s,  off, 32);
            s2 += __shfl_xor(s2, off, 32);
        }
        float mu  = s * (1.0f / 256.0f);
        float var = s2 * (1.0f / 256.0f) - mu * mu;
        float inv = rsqrtf(var + 1e-5f);
        int node = idx[rb + rr];
#pragma unroll
        for (int j = 0; j < 8; ++j) {
            int c = lane + 32 * j;
            float v = sbuf[rr][c];
            E[(size_t)node * 256 + c] = (v - mu) * inv * gamma[c] + beta[c];
        }
    }
}

// ---------- out[i,:] = E[lvl0_idx[i], :] ----------
__global__ void emit_kernel(const float* __restrict__ E,
                            const int* __restrict__ idx,
                            float* __restrict__ out) {
    int i = blockIdx.x;
    int d = threadIdx.x;
    out[(size_t)i * 256 + d] = E[(size_t)idx[i] * 256 + d];
}

extern "C" void kernel_launch(void* const* d_in, const int* in_sizes, int n_in,
                              void* d_out, int out_size, void* d_ws, size_t ws_size,
                              hipStream_t stream) {
    const int*   comp_ids   = (const int*)d_in[0];
    const int*   op_ids     = (const int*)d_in[1];
    const int*   lc         = (const int*)d_in[2];
    const int*   rc         = (const int*)d_in[3];
    const int*   tc         = (const int*)d_in[4];
    const int*   lvl2       = (const int*)d_in[5];
    const int*   lvl1       = (const int*)d_in[6];
    const int*   lvl0       = (const int*)d_in[7];
    const float* comp_table = (const float*)d_in[8];
    const float* op_table   = (const float*)d_in[9];
    const float* W1b        = (const float*)d_in[10];
    const float* b1b        = (const float*)d_in[11];
    const float* W2b        = (const float*)d_in[12];
    const float* b2b        = (const float*)d_in[13];
    const float* W1t        = (const float*)d_in[14];
    const float* b1t        = (const float*)d_in[15];
    const float* W2t        = (const float*)d_in[16];
    const float* b2t        = (const float*)d_in[17];
    const float* gamma      = (const float*)d_in[18];
    const float* beta       = (const float*)d_in[19];

    int N  = in_sizes[0];      // total nodes (B*T)
    int L2 = in_sizes[5];
    int L1 = in_sizes[6];
    int L0 = in_sizes[7];
    int Lmax = L2;

    // workspace layout
    char*  ws  = (char*)d_ws;
    size_t off = 0;
    auto alloc = [&](size_t bytes) -> void* {
        void* p = ws + off;
        off += (bytes + 255) & ~(size_t)255;
        return p;
    };
    float*          E    = (float*)         alloc((size_t)N * 256 * 4);
    unsigned short* X    = (unsigned short*)alloc((size_t)Lmax * 1024 * 2);
    unsigned short* Ab   = (unsigned short*)alloc((size_t)Lmax * 512 * 2);
    unsigned short* At   = (unsigned short*)alloc((size_t)Lmax * 512 * 2);
    float*          Rb   = (float*)         alloc((size_t)Lmax * 256 * 4);
    float*          Rt   = (float*)         alloc((size_t)Lmax * 256 * 4);
    int*            H3   = (int*)           alloc((size_t)Lmax * 4);
    unsigned short* W1bT = (unsigned short*)alloc((size_t)512 * 768 * 2);
    unsigned short* W1tT = (unsigned short*)alloc((size_t)512 * 1024 * 2);
    unsigned short* W2bT = (unsigned short*)alloc((size_t)256 * 512 * 2);
    unsigned short* W2tT = (unsigned short*)alloc((size_t)256 * 512 * 2);
    (void)ws_size;

    // initial embeddings
    init_embeds_kernel<<<N, 256, 0, stream>>>(E, comp_table, comp_ids);

    // weight conversion / transpose to bf16 [N,K]
    cvt_transpose_kernel<<<(768 * 512 + 255) / 256, 256, 0, stream>>>(W1b, W1bT, 768, 512);
    cvt_transpose_kernel<<<(1024 * 512 + 255) / 256, 256, 0, stream>>>(W1t, W1tT, 1024, 512);
    cvt_transpose_kernel<<<(512 * 256 + 255) / 256, 256, 0, stream>>>(W2b, W2bT, 512, 256);
    cvt_transpose_kernel<<<(512 * 256 + 255) / 256, 256, 0, stream>>>(W2t, W2tT, 512, 256);

    const int* idxs[3] = { lvl2, lvl1, lvl0 };
    int        Ls[3]   = { L2, L1, L0 };

    for (int lv = 0; lv < 3; ++lv) {
        const int* idx = idxs[lv];
        int L = Ls[lv];

        gather_kernel<<<L, 256, 0, stream>>>(E, op_table, op_ids, lc, rc, tc, idx,
                                             X, Rb, Rt, H3);

        mlp1_kernel<<<L / 32, 256, 0, stream>>>(X, 1024, 768, W1bT, b1b, Ab);
        mlp1_kernel<<<L / 32, 256, 0, stream>>>(X, 1024, 1024, W1tT, b1t, At);

        mlp2_kernel<<<L / 16, 256, 0, stream>>>(Ab, At, W2bT, W2tT, b2b, b2t,
                                                Rb, Rt, H3, gamma, beta, idx, E);
    }

    emit_kernel<<<L0, 256, 0, stream>>>(E, lvl0, (float*)d_out);
}
